// LSTMModel_69097433858209
// MI455X (gfx1250) — matrix operand order repaired and checked by reading, not compiled, not run on hardware
//
#include <hip/hip_runtime.h>

typedef _Float16 v16h __attribute__((ext_vector_type(16)));
typedef _Float16 v8h  __attribute__((ext_vector_type(8)));
typedef _Float16 v4h  __attribute__((ext_vector_type(4)));
typedef float    v8f  __attribute__((ext_vector_type(8)));

#define B_TOT 4096
#define T_LEN 512
#define IN_D  4
#define H_D   64
#define G4    256   // 4*H
#define KX    32    // padded K for the x/bias GEMM

// Fast activations: single v_exp_f32 + single v_rcp_f32 (hardware rcp, ~1 ulp).
__device__ __forceinline__ float sigf(float x) {
    return __builtin_amdgcn_rcpf(1.0f + __expf(-x));
}
__device__ __forceinline__ float tanh_fast(float x) {
    float e = __expf(2.0f * x);
    return 1.0f - 2.0f * __builtin_amdgcn_rcpf(e + 1.0f);
}

// A-fragment (16x32 f16, MxK) from an LDS buffer [16 rows x `stride` cols].
// Lanes 0-15 -> M=lane, K = {0..15}; lanes 16-31 -> M=lane-16, K = {16..31}
// (plus k0 offset). Each lane: two contiguous 16-byte runs -> ds_load_b128 pair.
__device__ __forceinline__ v16h ld_afrag(const _Float16* hb, int lane, int k0, int stride) {
    int m  = lane & 15;
    int kb = (lane >> 4) * 16;
    const v8h* p0 = (const v8h*)(hb + m * stride + k0 + kb);
    const v8h* p1 = (const v8h*)(hb + m * stride + k0 + kb + 8);
    v8h lo = *p0;
    v8h hi = *p1;
    v16h a;
#pragma unroll
    for (int j = 0; j < 8; ++j) { a[j] = lo[j]; a[8 + j] = hi[j]; }
    return a;
}

// B-fragment (32x16 f16, KxN) of W^T where W is row-major [256 x `stride`] f16 in LDS.
// Lane l holds column n0+(l&15), 16 contiguous K values starting at k0 + (l>=16 ? 16 : 0):
// one contiguous 32-byte LDS load per lane.
__device__ __forceinline__ v16h ld_bfrag(const _Float16* W, int lane, int n0, int k0, int stride) {
    int n  = n0 + (lane & 15);
    int ko = k0 + ((lane >> 4) << 4);
    return *(const v16h*)(W + n * stride + ko);
}

__device__ __forceinline__ v8f wmma_f16(v16h a, v16h b, v8f c) {
    return __builtin_amdgcn_wmma_f32_16x16x32_f16(false, a, false, b, (short)0, c, false, false);
}

// One fused timestep for both LSTM layers. h1/h2 are double-buffered (rd/wr),
// so only 2 barriers per step are needed (visibility only, no WAR barriers).
__device__ __forceinline__ void lstm_step(
    int t, int tid, int wave, int lane, int l15, int rowoff, int col,
    const float* __restrict__ xbase,
    const _Float16* __restrict__ h1_rd, _Float16* __restrict__ h1_wr,
    const _Float16* __restrict__ h2_rd, _Float16* __restrict__ h2_wr,
    _Float16* __restrict__ sXh,
    const _Float16* __restrict__ sWx0, const _Float16* __restrict__ sWh0,
    const _Float16* __restrict__ sWi1, const _Float16* __restrict__ sWh1,
    v8f& c1, v8f& c2, float4& xr,
    float b1i, float b1f, float b1g, float b1o)
{
    // stage x_t (value was prefetched one iteration ago)
    if (tid < 16) {
        v4h xh;
        xh[0] = (_Float16)xr.x; xh[1] = (_Float16)xr.y;
        xh[2] = (_Float16)xr.z; xh[3] = (_Float16)xr.w;
        *(v4h*)(sXh + tid * KX) = xh;
    }
    __syncthreads();   // s1: sXh + h1_rd/h2_rd (written last step) visible

    // prefetch x_{t+1}: full iteration of compute hides the HBM latency
    if (tid < 16 && (t + 1) < T_LEN)
        xr = ((const float4*)xbase)[(size_t)tid * T_LEN + (t + 1)];

    v16h ax    = ld_afrag(sXh,  lane, 0,  KX);
    v16h a0    = ld_afrag(h1_rd, lane, 0,  H_D);
    v16h a1    = ld_afrag(h1_rd, lane, 32, H_D);
    v16h ah2_0 = ld_afrag(h2_rd, lane, 0,  H_D);
    v16h ah2_1 = ld_afrag(h2_rd, lane, 32, H_D);

    // ===== layer 0 : gates = [x_t|1] @ [Wx0|b0]^T + h1_{t-1} @ Wh0^T =====
    v8f acc[4];
#pragma unroll
    for (int jt = 0; jt < 4; ++jt) {               // jt = gate (i,f,g,o)
        int n0 = jt * H_D + wave * 16;
        v8f z;
#pragma unroll
        for (int r = 0; r < 8; ++r) z[r] = 0.0f;   // inline-0 SRC2
        acc[jt] = wmma_f16(ax, ld_bfrag(sWx0, lane, n0, 0,  KX),  z);
        acc[jt] = wmma_f16(a0, ld_bfrag(sWh0, lane, n0, 0,  H_D), acc[jt]);
        acc[jt] = wmma_f16(a1, ld_bfrag(sWh0, lane, n0, 32, H_D), acc[jt]);
    }

    _Float16 hv1[8];
#pragma unroll
    for (int r = 0; r < 8; ++r) {
        float ig = sigf(acc[0][r]);
        float fg = sigf(acc[1][r]);
        float gg = tanh_fast(acc[2][r]);
        float og = sigf(acc[3][r]);
        float c  = fg * c1[r] + ig * gg;
        c1[r] = c;
        hv1[r] = (_Float16)(og * tanh_fast(c));
    }

    // ===== layer 1, independent half: h2_{t-1} @ Wh1^T (zero C, bias added later) =====
    v8f acc1[4];
#pragma unroll
    for (int jt = 0; jt < 4; ++jt) {
        int n0 = jt * H_D + wave * 16;
        v8f z;
#pragma unroll
        for (int r = 0; r < 8; ++r) z[r] = 0.0f;
        acc1[jt] = wmma_f16(ah2_0, ld_bfrag(sWh1, lane, n0, 0,  H_D), z);
        acc1[jt] = wmma_f16(ah2_1, ld_bfrag(sWh1, lane, n0, 32, H_D), acc1[jt]);
    }

    // publish h1_t into the *other* buffer (no WAR hazard -> no barrier needed before)
#pragma unroll
    for (int r = 0; r < 8; ++r)
        h1_wr[(r + rowoff) * H_D + col] = hv1[r];
    __syncthreads();   // s3: h1_t visible

    // ===== layer 1, dependent half: + h1_t @ Wi1^T =====
    v16h ah1_0 = ld_afrag(h1_wr, lane, 0,  H_D);
    v16h ah1_1 = ld_afrag(h1_wr, lane, 32, H_D);
#pragma unroll
    for (int jt = 0; jt < 4; ++jt) {
        int n0 = jt * H_D + wave * 16;
        acc1[jt] = wmma_f16(ah1_0, ld_bfrag(sWi1, lane, n0, 0,  H_D), acc1[jt]);
        acc1[jt] = wmma_f16(ah1_1, ld_bfrag(sWi1, lane, n0, 32, H_D), acc1[jt]);
    }

    _Float16 hv2[8];
#pragma unroll
    for (int r = 0; r < 8; ++r) {
        float ig = sigf(acc1[0][r] + b1i);
        float fg = sigf(acc1[1][r] + b1f);
        float gg = tanh_fast(acc1[2][r] + b1g);
        float og = sigf(acc1[3][r] + b1o);
        float c  = fg * c2[r] + ig * gg;
        c2[r] = c;
        hv2[r] = (_Float16)(og * tanh_fast(c));
    }
#pragma unroll
    for (int r = 0; r < 8; ++r)
        h2_wr[(r + rowoff) * H_D + col] = hv2[r];
    // visibility handled by next step's s1
}

__global__ __launch_bounds__(128) void lstm2_fused_kernel(
    const float* __restrict__ x,
    const float* __restrict__ W_ih0, const float* __restrict__ W_hh0,
    const float* __restrict__ b_ih0, const float* __restrict__ b_hh0,
    const float* __restrict__ W_ih1, const float* __restrict__ W_hh1,
    const float* __restrict__ b_ih1, const float* __restrict__ b_hh1,
    const float* __restrict__ W_fc,  const float* __restrict__ b_fc,
    float* __restrict__ out)
{
    __shared__ __align__(32) _Float16 sWh0[G4 * H_D];   // 32 KB  W_hh0 (f16)
    __shared__ __align__(32) _Float16 sWi1[G4 * H_D];   // 32 KB  W_ih1 (f16)
    __shared__ __align__(32) _Float16 sWh1[G4 * H_D];   // 32 KB  W_hh1 (f16)
    __shared__ __align__(32) _Float16 sWx0[G4 * KX];    // 16 KB  [W_ih0 | bias0 | 0...]
    __shared__ __align__(32) _Float16 sH1a[16 * H_D];   // h1 double buffer
    __shared__ __align__(32) _Float16 sH1b[16 * H_D];
    __shared__ __align__(32) _Float16 sH2a[16 * H_D];   // h2 double buffer
    __shared__ __align__(32) _Float16 sH2b[16 * H_D];
    __shared__ __align__(32) _Float16 sXh[16 * KX];     // [x_t | 1.0 | 0...]
    __shared__ float sB1[G4];                           // b_ih1 + b_hh1
    __shared__ float sWfc[H_D];

    const int tid  = threadIdx.x;
    const int wave = tid >> 5;            // 0..3 : owns h-columns [16*wave, 16*wave+16)
    const int lane = tid & 31;
    const int l15  = lane & 15;
    const int rowoff = (lane >> 4) * 8;   // C-layout: row m = r + rowoff
    const int col  = wave * 16 + l15;     // h-column this lane handles in C layout
    const int b0   = blockIdx.x * 16;     // batch tile base

    // ---- stage weights / tables into LDS (f32 -> f16 convert once) ----
    for (int i = tid; i < G4 * H_D; i += 128) {
        sWh0[i] = (_Float16)W_hh0[i];
        sWi1[i] = (_Float16)W_ih1[i];
        sWh1[i] = (_Float16)W_hh1[i];
    }
    for (int i = tid; i < G4 * KX; i += 128) sWx0[i] = (_Float16)0.0f;
    for (int i = tid; i < 16 * KX; i += 128) sXh[i]  = (_Float16)0.0f;
    __syncthreads();   // zero-fill before sparse writes below
    for (int i = tid; i < G4; i += 128) {
        sWx0[i * KX + 0] = (_Float16)W_ih0[i * IN_D + 0];
        sWx0[i * KX + 1] = (_Float16)W_ih0[i * IN_D + 1];
        sWx0[i * KX + 2] = (_Float16)W_ih0[i * IN_D + 2];
        sWx0[i * KX + 3] = (_Float16)W_ih0[i * IN_D + 3];
        sWx0[i * KX + 4] = (_Float16)(b_ih0[i] + b_hh0[i]);   // bias via constant-1 input
        sB1[i] = b_ih1[i] + b_hh1[i];
    }
    if (tid < 16) sXh[tid * KX + 4] = (_Float16)1.0f;         // the constant-1 column
    if (tid < H_D) sWfc[tid] = W_fc[tid];
    for (int i = tid; i < 16 * H_D; i += 128) {
        sH1a[i] = (_Float16)0.0f;
        sH2a[i] = (_Float16)0.0f;
    }
    __syncthreads();   // init complete (sB1 readable)

    // loop-invariant layer-1 biases -> registers (one per gate for this lane's column)
    const float b1i = sB1[0 * H_D + col];
    const float b1f = sB1[1 * H_D + col];
    const float b1g = sB1[2 * H_D + col];
    const float b1o = sB1[3 * H_D + col];

    // Cell states, C-matrix layout (8 rows per lane for this wave's 16 columns).
    v8f c1, c2;
#pragma unroll
    for (int r = 0; r < 8; ++r) { c1[r] = 0.0f; c2[r] = 0.0f; }

    const float* xbase = x + (size_t)b0 * T_LEN * IN_D;

    // prefetch x_0
    float4 xr;
    if (tid < 16) xr = ((const float4*)xbase)[(size_t)tid * T_LEN];

    // unroll by 2: double-buffer swap is compile-time
    for (int t = 0; t < T_LEN; t += 2) {
        lstm_step(t,     tid, wave, lane, l15, rowoff, col, xbase,
                  sH1a, sH1b, sH2a, sH2b, sXh, sWx0, sWh0, sWi1, sWh1,
                  c1, c2, xr, b1i, b1f, b1g, b1o);
        lstm_step(t + 1, tid, wave, lane, l15, rowoff, col, xbase,
                  sH1b, sH1a, sH2b, sH2a, sXh, sWx0, sWh0, sWi1, sWh1,
                  c1, c2, xr, b1i, b1f, b1g, b1o);
    }

    __syncthreads();
    // ---- final head: out[b] = h2_last[b] . W_fc + b_fc  (t=511 wrote sH2a) ----
    if (tid < 16) {
        float sum = b_fc[0];
#pragma unroll
        for (int k = 0; k < H_D; ++k)
            sum += (float)sH2a[tid * H_D + k] * sWfc[k];
        out[b0 + tid] = sum;
    }
}

extern "C" void kernel_launch(void* const* d_in, const int* in_sizes, int n_in,
                              void* d_out, int out_size, void* d_ws, size_t ws_size,
                              hipStream_t stream) {
    (void)in_sizes; (void)n_in; (void)out_size; (void)d_ws; (void)ws_size;
    const float* x     = (const float*)d_in[0];
    const float* W_ih0 = (const float*)d_in[1];
    const float* W_hh0 = (const float*)d_in[2];
    const float* b_ih0 = (const float*)d_in[3];
    const float* b_hh0 = (const float*)d_in[4];
    const float* W_ih1 = (const float*)d_in[5];
    const float* W_hh1 = (const float*)d_in[6];
    const float* b_ih1 = (const float*)d_in[7];
    const float* b_hh1 = (const float*)d_in[8];
    const float* W_fc  = (const float*)d_in[9];
    const float* b_fc  = (const float*)d_in[10];
    float* out = (float*)d_out;

    dim3 grid(B_TOT / 16);   // 256 workgroups, each owns a 16-row batch tile
    dim3 block(128);         // 4 wave32s
    lstm2_fused_kernel<<<grid, block, 0, stream>>>(
        x, W_ih0, W_hh0, b_ih0, b_hh0,
        W_ih1, W_hh1, b_ih1, b_hh1, W_fc, b_fc, out);
}
